// LSMVOS_85143431676042
// MI455X (gfx1250) — matrix operand order repaired and checked by reading, not compiled, not run on hardware
//
#include <hip/hip_runtime.h>
#include <math.h>

// ---------------------------------------------------------------------------
// Types for CDNA5 WMMA
// ---------------------------------------------------------------------------
typedef _Float16 half_t;
typedef _Float16 v16h __attribute__((ext_vector_type(16)));
typedef _Float16 v8h  __attribute__((ext_vector_type(8)));
typedef float    v8f  __attribute__((ext_vector_type(8)));

#define NEG_INF (-__builtin_inff())

// Problem constants (from reference setup_inputs)
#define BN    2
#define H0    48
#define W0    48
#define P0    (H0*W0)          // 2304
#define CG    2304             // kg1 channels (H*W)
#define CL    289              // klp channels (17*17)
#define CCAT  1281             // 256*5 + 1

#define LSTRIDE 48             // LDS row stride in halves (96B, 16B-aligned)

// ---------------------------------------------------------------------------
// 8x8 adaptive max pool of key_mask / pre_mask; also writes pm into xcat ch1280
// ---------------------------------------------------------------------------
__global__ void pool8_kernel(const float* __restrict__ km, const float* __restrict__ pmm,
                             float* __restrict__ gm, float* __restrict__ lm,
                             half_t* __restrict__ xcat)
{
    int idx = blockIdx.x * blockDim.x + threadIdx.x;
    if (idx >= BN * P0) return;
    int b = idx / P0, i = idx % P0;
    int y = i / W0, x = i % W0;
    const float* k0 = km  + ((size_t)b * 384 + y * 8) * 384 + x * 8;
    const float* p0 = pmm + ((size_t)b * 384 + y * 8) * 384 + x * 8;
    float mk = NEG_INF, mp = NEG_INF;
    for (int dy = 0; dy < 8; ++dy)
        for (int dx = 0; dx < 8; ++dx) {
            mk = fmaxf(mk, k0[dy * 384 + dx]);
            mp = fmaxf(mp, p0[dy * 384 + dx]);
        }
    gm[idx] = mk;
    lm[idx] = mp;
    xcat[((size_t)(b * CCAT + 1280)) * P0 + i] = (half_t)mp;
}

// ---------------------------------------------------------------------------
// p3 (f32) -> xcat channels [0,256) (f16)
// ---------------------------------------------------------------------------
__global__ void copy_p3_kernel(const float* __restrict__ p3, half_t* __restrict__ xcat)
{
    int idx = blockIdx.x * blockDim.x + threadIdx.x;
    if (idx >= BN * 256 * P0) return;
    int b = idx / (256 * P0);
    int rem = idx - b * 256 * P0;
    int c = rem / P0, i = rem % P0;
    xcat[((size_t)(b * CCAT + c)) * P0 + i] = (half_t)p3[idx];
}

// ---------------------------------------------------------------------------
// weight convert + transpose + pad: src OIHW [Cout,Cin,3,3] f32 ->
// dst [Cout][r=ky*3+kx][ci_pad] f16, zero-filled for ci >= Cin
// ---------------------------------------------------------------------------
__global__ void cvt_w_kernel(const float* __restrict__ src, half_t* __restrict__ dst,
                             int Cout, int Cin, int Cin_pad)
{
    int idx = blockIdx.x * blockDim.x + threadIdx.x;
    int n = Cout * 9 * Cin_pad;
    if (idx >= n) return;
    int co = idx / (9 * Cin_pad);
    int rem = idx - co * 9 * Cin_pad;
    int r = rem / Cin_pad;
    int cp = rem - r * Cin_pad;
    dst[idx] = (cp < Cin) ? (half_t)src[((size_t)co * Cin + cp) * 9 + r] : (half_t)0.0f;
}

// ---------------------------------------------------------------------------
// descending bitonic sort in LDS (n power of two, all `nt` threads call)
// ---------------------------------------------------------------------------
__device__ inline void bitonic_desc(float* s, int n, int tid, int nt)
{
    for (int k = 2; k <= n; k <<= 1) {
        for (int j = k >> 1; j > 0; j >>= 1) {
            for (int i = tid; i < n; i += nt) {
                int ixj = i ^ j;
                if (ixj > i) {
                    bool dir = ((i & k) == 0);   // true -> descending run
                    float a = s[i], c = s[ixj];
                    if ((a < c) == dir) { s[i] = c; s[ixj] = a; }
                }
            }
            __syncthreads();
        }
    }
}

// ---------------------------------------------------------------------------
// global topk: per pixel sort kg1[:,pix]*gm (fg) and *(1-gm) (bg), keep 256 desc
// -> xcat channels [256,512) and [512,768)
// ---------------------------------------------------------------------------
__global__ __launch_bounds__(256)
void topk_global_kernel(const float* __restrict__ kg1, const float* __restrict__ gm,
                        half_t* __restrict__ xcat)
{
    __shared__ float vals[CG];
    __shared__ float srt[4096];
    const int pix = blockIdx.x, b = blockIdx.y, tid = threadIdx.x;
    for (int c = tid; c < CG; c += 256)
        vals[c] = kg1[((size_t)(b * CG + c)) * P0 + pix];
    __syncthreads();
    for (int pass = 0; pass < 2; ++pass) {
        for (int c = tid; c < 4096; c += 256) {
            float v = NEG_INF;
            if (c < CG) {
                float m = gm[b * CG + c];
                v = vals[c] * (pass == 0 ? m : 1.0f - m);
            }
            srt[c] = v;
        }
        __syncthreads();
        bitonic_desc(srt, 4096, tid, 256);
        int cb = 256 + pass * 256;
        xcat[((size_t)(b * CCAT + cb + tid)) * P0 + pix] = (half_t)srt[tid];
        __syncthreads();
    }
}

// ---------------------------------------------------------------------------
// local topk: patches of pooled pre_mask (17x17 zero-padded window), klp*patch
// -> xcat channels [768,1024) (fg) and [1024,1280) (bg)
// ---------------------------------------------------------------------------
__global__ __launch_bounds__(256)
void topk_local_kernel(const float* __restrict__ klp, const float* __restrict__ lm,
                       half_t* __restrict__ xcat)
{
    __shared__ float vals[CL];
    __shared__ float patch[CL];
    __shared__ float srt[512];
    const int pix = blockIdx.x, b = blockIdx.y, tid = threadIdx.x;
    const int y = pix / W0, x = pix % W0;
    for (int c = tid; c < CL; c += 256) {
        vals[c] = klp[((size_t)(b * CL + c)) * P0 + pix];
        int dy = c / 17, dx = c % 17;
        int iy = y + dy - 8, ix = x + dx - 8;
        patch[c] = (iy >= 0 && iy < H0 && ix >= 0 && ix < W0) ? lm[b * P0 + iy * W0 + ix] : 0.0f;
    }
    __syncthreads();
    for (int pass = 0; pass < 2; ++pass) {
        for (int c = tid; c < 512; c += 256)
            srt[c] = (c < CL) ? vals[c] * (pass == 0 ? patch[c] : 1.0f - patch[c]) : NEG_INF;
        __syncthreads();
        bitonic_desc(srt, 512, tid, 256);
        int cb = 768 + pass * 256;
        xcat[((size_t)(b * CCAT + cb + tid)) * P0 + pix] = (half_t)srt[tid];
        __syncthreads();
    }
}

// ---------------------------------------------------------------------------
// K-chunk staging for the implicit-GEMM conv; all shape params compile-time so
// the 32 activation loads become one base address + immediate offsets.
// ---------------------------------------------------------------------------
template<int CIN, int CPAD, int COUT, int HH, int WW>
__device__ __forceinline__ void stage_chunk(int t, _Float16* lA, _Float16* lB,
                                            const half_t* __restrict__ w2,
                                            const half_t* __restrict__ inB,
                                            int co0, int tid, int py, int px)
{
    constexpr int P   = HH * WW;
    constexpr int CHK = CPAD / 32;
    constexpr int K2  = 9 * CPAD;
    const int rr = t / CHK;
    const int cb = (t - rr * CHK) * 32;
    const int ky = rr / 3;
    const int kx = rr - ky * 3;
    const int iy = py + ky - 1;
    const int ix = px + kx - 1;
    const bool rv = (iy >= 0) & (iy < HH) & (ix >= 0) & (ix < WW);
    const int rowoff = rv ? (iy * WW + ix) : 0;
    const half_t* __restrict__ q = inB + rowoff + (size_t)cb * P;
    // A: 32 co x 32 k, coalesced over kk (pre-padded with zeros -> no guard)
#pragma unroll
    for (int u = 0; u < 8; ++u) {
        int e  = tid + u * 128;
        int m  = e >> 5;
        int kk = e & 31;
        lA[m * LSTRIDE + kk] = w2[(size_t)(co0 + m) * K2 + rr * CPAD + cb + kk];
    }
    // B: this thread's pixel, 32 channels at immediate offsets kk*P
#pragma unroll
    for (int kk = 0; kk < 32; ++kk) {
        half_t v = q[(size_t)kk * P];            // clamped-safe address
        bool ok = rv;
        if constexpr (CPAD != CIN) ok = ok & ((cb + kk) < CIN);
        lB[tid * LSTRIDE + kk] = ok ? v : (half_t)0.0f;
    }
}

// ---------------------------------------------------------------------------
// Implicit-GEMM 3x3 conv (pad 1) via v_wmma_f32_16x16x32_f16.
// in  : [B, Cin, H, W]  f16 ; w2: [Cout][r][ci_pad] f16 ; bias f32
// out : [B, Cout, H, W] f16
// Block = 128 threads (4 wave32). Block tile: 32 outC x 128 pixels.
// Each wave: 32 outC x 32 pixels = 4 WMMA per K-chunk.
// LDS double-buffered -> single barrier per chunk.
// Grid: (P/128, Cout/32, B).
// ---------------------------------------------------------------------------
template<int CIN, int CPAD, int COUT, int HH, int WW>
__global__ __launch_bounds__(128)
void conv3x3_wmma_kernel(const half_t* __restrict__ in, const half_t* __restrict__ w2,
                         const float* __restrict__ bias, half_t* __restrict__ out)
{
    constexpr int P   = HH * WW;
    constexpr int ASZ = 32 * LSTRIDE;
    constexpr int BSZ = 128 * LSTRIDE;
    constexpr int NCH = 9 * (CPAD / 32);

    __shared__ __align__(16) _Float16 ldsA[2 * ASZ];
    __shared__ __align__(16) _Float16 ldsB[2 * BSZ];

    const int b   = blockIdx.z;
    const int co0 = blockIdx.y * 32;
    const int pixBase = blockIdx.x * 128;

    const int tid  = threadIdx.x;
    const int lane = tid & 31;
    const int wave = tid >> 5;
    const int hi   = (lane >> 4) & 1;
    const int mrow = lane & 15;

    const int pj = pixBase + tid;
    const int py = pj / WW;
    const int px = pj % WW;
    const half_t* __restrict__ inB = in + (size_t)b * CIN * P;

    v8f acc00 = {}, acc01 = {}, acc10 = {}, acc11 = {};

    const int aRow0 = (mrow)      * LSTRIDE + hi * 8;
    const int aRow1 = (16 + mrow) * LSTRIDE + hi * 8;
    const int bRow0 = (wave * 32 + mrow)      * LSTRIDE + hi * 8;
    const int bRow1 = (wave * 32 + 16 + mrow) * LSTRIDE + hi * 8;

    stage_chunk<CIN, CPAD, COUT, HH, WW>(0, ldsA, ldsB, w2, inB, co0, tid, py, px);

    for (int t = 0; t < NCH; ++t) {
        __syncthreads();
        const _Float16* lA = ldsA + (t & 1) * ASZ;
        const _Float16* lB = ldsB + (t & 1) * BSZ;

        v8h a00 = *(const v8h*)&lA[aRow0];
        v8h a01 = *(const v8h*)&lA[aRow0 + 16];
        v8h a10 = *(const v8h*)&lA[aRow1];
        v8h a11 = *(const v8h*)&lA[aRow1 + 16];
        v8h b00 = *(const v8h*)&lB[bRow0];
        v8h b01 = *(const v8h*)&lB[bRow0 + 16];
        v8h b10 = *(const v8h*)&lB[bRow1];
        v8h b11 = *(const v8h*)&lB[bRow1 + 16];
        v16h af0, af1, bf0, bf1;
#pragma unroll
        for (int i = 0; i < 8; ++i) {
            af0[i] = a00[i]; af0[i + 8] = a01[i];
            af1[i] = a10[i]; af1[i + 8] = a11[i];
            bf0[i] = b00[i]; bf0[i + 8] = b01[i];
            bf1[i] = b10[i]; bf1[i + 8] = b11[i];
        }
        acc00 = __builtin_amdgcn_wmma_f32_16x16x32_f16(false, af0, false, bf0, (short)0, acc00, false, false);
        acc01 = __builtin_amdgcn_wmma_f32_16x16x32_f16(false, af0, false, bf1, (short)0, acc01, false, false);
        acc10 = __builtin_amdgcn_wmma_f32_16x16x32_f16(false, af1, false, bf0, (short)0, acc10, false, false);
        acc11 = __builtin_amdgcn_wmma_f32_16x16x32_f16(false, af1, false, bf1, (short)0, acc11, false, false);

        if (t + 1 < NCH)
            stage_chunk<CIN, CPAD, COUT, HH, WW>(t + 1,
                ldsA + ((t + 1) & 1) * ASZ, ldsB + ((t + 1) & 1) * BSZ,
                w2, inB, co0, tid, py, px);
    }

    // D layout: vgpr v, lane l: M = v + (l>=16 ? 8 : 0), N = l&15
    const int pix0 = pixBase + wave * 32 + mrow;
    const int pix1 = pix0 + 16;
#pragma unroll
    for (int v = 0; v < 8; ++v) {
        int coA = co0 + v + hi * 8;
        int coB = coA + 16;
        float bA = bias[coA], bB = bias[coB];
        out[((size_t)(b * COUT + coA)) * P + pix0] = (half_t)(acc00[v] + bA);
        out[((size_t)(b * COUT + coA)) * P + pix1] = (half_t)(acc01[v] + bA);
        out[((size_t)(b * COUT + coB)) * P + pix0] = (half_t)(acc10[v] + bB);
        out[((size_t)(b * COUT + coB)) * P + pix1] = (half_t)(acc11[v] + bB);
    }
}

template<int CIN, int CPAD, int COUT, int HH, int WW>
static void launch_conv(const half_t* in, const half_t* w, const float* bias,
                        half_t* out, hipStream_t stream)
{
    constexpr int P = HH * WW;
    conv3x3_wmma_kernel<CIN, CPAD, COUT, HH, WW>
        <<<dim3(P / 128, COUT / 32, BN), 128, 0, stream>>>(in, w, bias, out);
}

// ---------------------------------------------------------------------------
// InstanceNorm (affine=False, biased var, eps=1e-5) + ReLU, in place, f16.
// grid = B*C, block = 256
// ---------------------------------------------------------------------------
__global__ __launch_bounds__(256)
void inorm_relu_kernel(half_t* __restrict__ x, int P)
{
    half_t* p = x + (size_t)blockIdx.x * P;
    float s = 0.0f, sq = 0.0f;
    for (int i = threadIdx.x; i < P; i += 256) {
        float v = (float)p[i];
        s += v; sq += v * v;
    }
    __shared__ float rs[256], rq[256];
    rs[threadIdx.x] = s; rq[threadIdx.x] = sq;
    __syncthreads();
    for (int o = 128; o > 0; o >>= 1) {
        if (threadIdx.x < o) { rs[threadIdx.x] += rs[threadIdx.x + o]; rq[threadIdx.x] += rq[threadIdx.x + o]; }
        __syncthreads();
    }
    float invP = 1.0f / (float)P;
    float mean = rs[0] * invP;
    float var  = fmaxf(rq[0] * invP - mean * mean, 0.0f);
    float inv  = rsqrtf(var + 1e-5f);
    for (int i = threadIdx.x; i < P; i += 256) {
        float v = ((float)p[i] - mean) * inv;
        p[i] = (half_t)(v > 0.0f ? v : 0.0f);
    }
}

// ---------------------------------------------------------------------------
// bilinear resize, align_corners=True, f16 -> f16. CH = B*C planes.
// ---------------------------------------------------------------------------
__global__ void resize_ac_kernel(const half_t* __restrict__ in, half_t* __restrict__ out,
                                 int CH, int hin, int win, int hout, int wout)
{
    int idx = blockIdx.x * blockDim.x + threadIdx.x;
    if (idx >= CH * hout * wout) return;
    int bc = idx / (hout * wout);
    int rem = idx - bc * (hout * wout);
    int Y = rem / wout, X = rem % wout;
    float sy = (float)(hin - 1) / (float)(hout - 1);
    float sx = (float)(win - 1) / (float)(wout - 1);
    float ys = Y * sy, xs = X * sx;
    int y0 = (int)floorf(ys); if (y0 < 0) y0 = 0; if (y0 > hin - 1) y0 = hin - 1;
    int x0 = (int)floorf(xs); if (x0 < 0) x0 = 0; if (x0 > win - 1) x0 = win - 1;
    int y1 = (y0 + 1 < hin) ? y0 + 1 : hin - 1;
    int x1 = (x0 + 1 < win) ? x0 + 1 : win - 1;
    float wy = ys - (float)y0, wx = xs - (float)x0;
    const half_t* pl = in + (size_t)bc * hin * win;
    float v00 = (float)pl[y0 * win + x0], v01 = (float)pl[y0 * win + x1];
    float v10 = (float)pl[y1 * win + x0], v11 = (float)pl[y1 * win + x1];
    float top = v00 * (1.0f - wx) + v01 * wx;
    float bot = v10 * (1.0f - wx) + v11 * wx;
    out[idx] = (half_t)(top * (1.0f - wy) + bot * wy);
}

// ---------------------------------------------------------------------------
// residual add: o = f16( f16(a) + f32(b) )
// ---------------------------------------------------------------------------
__global__ void add_f16f32_kernel(const half_t* __restrict__ a, const float* __restrict__ b,
                                  half_t* __restrict__ o, int n)
{
    int i = blockIdx.x * blockDim.x + threadIdx.x;
    if (i < n) o[i] = (half_t)((float)a[i] + b[i]);
}

// ---------------------------------------------------------------------------
// pred: 3x3 conv 64->1 + sigmoid, direct (tiny), f16 in, f32 out
// ---------------------------------------------------------------------------
__global__ void pred_sigmoid_kernel(const half_t* __restrict__ z, const float* __restrict__ w,
                                    const float* __restrict__ bias, float* __restrict__ out)
{
    int idx = blockIdx.x * blockDim.x + threadIdx.x;
    const int Hh = 192, Ww = 192, Pp = Hh * Ww;
    if (idx >= BN * Pp) return;
    int b = idx / Pp, pix = idx % Pp;
    int y = pix / Ww, x = pix % Ww;
    float acc = bias[0];
    for (int ci = 0; ci < 64; ++ci) {
        const half_t* pl = z + (size_t)(b * 64 + ci) * Pp;
        const float* wl = w + ci * 9;
        for (int ky = 0; ky < 3; ++ky) {
            int iy = y + ky - 1;
            if (iy < 0 || iy >= Hh) continue;
            for (int kx = 0; kx < 3; ++kx) {
                int ix = x + kx - 1;
                if (ix < 0 || ix >= Ww) continue;
                acc += (float)pl[iy * Ww + ix] * wl[ky * 3 + kx];
            }
        }
    }
    out[idx] = 1.0f / (1.0f + expf(-acc));
}

// ---------------------------------------------------------------------------
// host-side launch
// ---------------------------------------------------------------------------
extern "C" void kernel_launch(void* const* d_in, const int* in_sizes, int n_in,
                              void* d_out, int out_size, void* d_ws, size_t ws_size,
                              hipStream_t stream)
{
    (void)in_sizes; (void)n_in; (void)out_size; (void)ws_size;

    const float* p3       = (const float*)d_in[0];
    const float* p2       = (const float*)d_in[1];
    const float* p1       = (const float*)d_in[2];
    const float* kg1      = (const float*)d_in[3];
    const float* klp      = (const float*)d_in[4];
    const float* key_mask = (const float*)d_in[5];
    const float* pre_mask = (const float*)d_in[6];

    // workspace bump allocator
    size_t off = 0;
    auto alloc = [&](size_t bytes) -> void* {
        void* p = (char*)d_ws + off;
        off = (off + bytes + 255) & ~(size_t)255;
        return p;
    };
    const int CT_PAD = 1312;   // ceil(1281/32)*32
    float*  gm   = (float*)alloc(BN * P0 * 4);
    float*  lm   = (float*)alloc(BN * P0 * 4);
    half_t* xcat = (half_t*)alloc((size_t)BN * CCAT * P0 * 2);

    half_t* w_ct   = (half_t*)alloc((size_t)256 * 9 * CT_PAD * 2);
    half_t* w_a1   = (half_t*)alloc(64 * 9 * 256 * 2);
    half_t* w_a2   = (half_t*)alloc(256 * 9 * 64 * 2);
    half_t* w_r1a1 = (half_t*)alloc(64 * 9 * 256 * 2);
    half_t* w_r1a2 = (half_t*)alloc(256 * 9 * 64 * 2);
    half_t* w_r1b1 = (half_t*)alloc(64 * 9 * 256 * 2);
    half_t* w_r1b2 = (half_t*)alloc(256 * 9 * 64 * 2);
    half_t* w_dc   = (half_t*)alloc(64 * 9 * 256 * 2);
    half_t* w_r2a1 = (half_t*)alloc(64 * 9 * 64 * 2);
    half_t* w_r2a2 = (half_t*)alloc(64 * 9 * 64 * 2);
    half_t* w_r2b1 = (half_t*)alloc(64 * 9 * 64 * 2);
    half_t* w_r2b2 = (half_t*)alloc(64 * 9 * 64 * 2);

    half_t* t0   = (half_t*)alloc((size_t)BN * 256 * P0 * 2);
    half_t* h48  = (half_t*)alloc((size_t)BN * 64 * P0 * 2);
    half_t* x1   = (half_t*)alloc((size_t)BN * 256 * P0 * 2);
    half_t* yA   = (half_t*)alloc((size_t)BN * 256 * 9216 * 2);
    half_t* yB   = (half_t*)alloc((size_t)BN * 256 * 9216 * 2);
    half_t* h96  = (half_t*)alloc((size_t)BN * 64 * 9216 * 2);
    half_t* d96  = (half_t*)alloc((size_t)BN * 64 * 9216 * 2);
    half_t* zA   = (half_t*)alloc((size_t)BN * 64 * 36864 * 2);
    half_t* zB   = (half_t*)alloc((size_t)BN * 64 * 36864 * 2);
    half_t* h192 = (half_t*)alloc((size_t)BN * 64 * 36864 * 2);

    // 1) pools + feature concat pieces
    pool8_kernel<<<dim3((BN * P0 + 255) / 256), 256, 0, stream>>>(key_mask, pre_mask, gm, lm, xcat);
    copy_p3_kernel<<<dim3((BN * 256 * P0 + 255) / 256), 256, 0, stream>>>(p3, xcat);
    topk_global_kernel<<<dim3(P0, BN), 256, 0, stream>>>(kg1, gm, xcat);
    topk_local_kernel<<<dim3(P0, BN), 256, 0, stream>>>(klp, lm, xcat);

    // 2) weight conversions (f32 OIHW -> f16 [Cout][r][ci_pad])
    struct WC { int src; half_t* dst; int Cout, Cin, Cpad; } wcs[] = {
        {7,  w_ct,   256, 1281, CT_PAD},
        {9,  w_a1,   64, 256, 256},  {11, w_a2,   256, 64, 64},
        {13, w_r1a1, 64, 256, 256},  {15, w_r1a2, 256, 64, 64},
        {17, w_r1b1, 64, 256, 256},  {19, w_r1b2, 256, 64, 64},
        {29, w_dc,   64, 256, 256},
        {21, w_r2a1, 64, 64, 64},    {23, w_r2a2, 64, 64, 64},
        {25, w_r2b1, 64, 64, 64},    {27, w_r2b2, 64, 64, 64},
    };
    for (auto& wc : wcs) {
        int n = wc.Cout * 9 * wc.Cpad;
        cvt_w_kernel<<<dim3((n + 255) / 256), 256, 0, stream>>>(
            (const float*)d_in[wc.src], wc.dst, wc.Cout, wc.Cin, wc.Cpad);
    }

    auto inorm = [&](half_t* x, int C, int P) {
        inorm_relu_kernel<<<dim3(BN * C), 256, 0, stream>>>(x, P);
    };
    auto bias = [&](int i) { return (const float*)d_in[i]; };

    // 3) conv_top + aic_top at 48x48
    launch_conv<1281, 1312, 256, 48, 48>(xcat, w_ct, bias(8), t0, stream);
    launch_conv< 256,  256,  64, 48, 48>(t0, w_a1, bias(10), h48, stream);  inorm(h48, 64, P0);
    launch_conv<  64,   64, 256, 48, 48>(h48, w_a2, bias(12), x1, stream);  inorm(x1, 256, P0);

    // 4) 48 -> 96, Refine1
    resize_ac_kernel<<<dim3((BN * 256 * 9216 + 255) / 256), 256, 0, stream>>>(
        x1, yA, BN * 256, 48, 48, 96, 96);
    launch_conv<256, 256,  64, 96, 96>(yA, w_r1a1, bias(14), h96, stream);  inorm(h96, 64, 9216);
    launch_conv< 64,  64, 256, 96, 96>(h96, w_r1a2, bias(16), yB, stream);  inorm(yB, 256, 9216);
    add_f16f32_kernel<<<dim3((BN * 256 * 9216 + 255) / 256), 256, 0, stream>>>(
        yB, p2, yA, BN * 256 * 9216);
    launch_conv<256, 256,  64, 96, 96>(yA, w_r1b1, bias(18), h96, stream);  inorm(h96, 64, 9216);
    launch_conv< 64,  64, 256, 96, 96>(h96, w_r1b2, bias(20), yB, stream);  inorm(yB, 256, 9216);
    launch_conv<256, 256,  64, 96, 96>(yB, w_dc, bias(30), d96, stream);    // dec_conv1, no norm

    // 5) 96 -> 192, Refine2
    resize_ac_kernel<<<dim3((BN * 64 * 36864 + 255) / 256), 256, 0, stream>>>(
        d96, zA, BN * 64, 96, 96, 192, 192);
    launch_conv<64, 64, 64, 192, 192>(zA, w_r2a1, bias(22), h192, stream);  inorm(h192, 64, 36864);
    launch_conv<64, 64, 64, 192, 192>(h192, w_r2a2, bias(24), zB, stream);  inorm(zB, 64, 36864);
    add_f16f32_kernel<<<dim3((BN * 64 * 36864 + 255) / 256), 256, 0, stream>>>(
        zB, p1, zA, BN * 64 * 36864);
    launch_conv<64, 64, 64, 192, 192>(zA, w_r2b1, bias(26), h192, stream);  inorm(h192, 64, 36864);
    launch_conv<64, 64, 64, 192, 192>(h192, w_r2b2, bias(28), zB, stream);  inorm(zB, 64, 36864);

    // 6) pred + sigmoid -> d_out (f32, [2,1,192,192])
    pred_sigmoid_kernel<<<dim3((BN * 36864 + 255) / 256), 256, 0, stream>>>(
        zB, (const float*)d_in[31], (const float*)d_in[32], (float*)d_out);
}
